// TaskAlignedAssigner_4252017623305
// MI455X (gfx1250) — compile-verified, matrix operand order checked by reading, not run on hardware
//
#include <hip/hip_runtime.h>
#include <hip/hip_bf16.h>

#define BS   16
#define NA   8400
#define NC   80
#define NGT  64
#define TOPKK 13
#define EPSF 1e-9f
#define USE_ASYNC_LDS 1

typedef __attribute__((ext_vector_type(16))) _Float16 v16h;
typedef __attribute__((ext_vector_type(8)))  float    v8f;

// ---------------------------------------------------------------------------
// Kernel 1: per (b, anchor) compute, for all 64 GTs:
//   overlaps (poly-AABB IoU), mask_in_gts (point-in-quad), align metric.
// GT polys staged into LDS with CDNA5 async-to-LDS loads.
// ---------------------------------------------------------------------------
__global__ void k_metrics(const float* __restrict__ pd_scores,
                          const float* __restrict__ pd_bboxes,
                          const float* __restrict__ anc,
                          const int*   __restrict__ gt_labels,
                          const float* __restrict__ gt_bboxes,
                          float* __restrict__ ov,
                          float* __restrict__ met,
                          unsigned char* __restrict__ maskin) {
  __shared__ float sh_raw[NGT * 9];
  __shared__ float sgx1[NGT], sgx2[NGT], sgy1[NGT], sgy2[NGT], sarea[NGT];
  __shared__ float scx[NGT * 4], scy[NGT * 4];
  __shared__ int   slabel[NGT];

  const int b = blockIdx.y;
  const int t = threadIdx.x;
  const float* gtb = gt_bboxes + (size_t)b * NGT * 9;

#if USE_ASYNC_LDS
  for (int i = t; i < NGT * 9; i += 256) {
    unsigned ldsa = (unsigned)(size_t)&sh_raw[i];   // low 32 bits of flat LDS addr = LDS offset
    const float* gp = gtb + i;
    asm volatile("global_load_async_to_lds_b32 %0, %1, off"
                 :: "v"(ldsa), "v"(gp) : "memory");
  }
  asm volatile("s_wait_asynccnt 0" ::: "memory");
#else
  for (int i = t; i < NGT * 9; i += 256) sh_raw[i] = gtb[i];
#endif
  __syncthreads();

  if (t < NGT) {
    const float* p = &sh_raw[t * 9];
    float x0 = p[0], y0 = p[1], x1 = p[2], y1 = p[3];
    float x2 = p[4], y2 = p[5], x3 = p[6], y3 = p[7];
    float gx1 = fminf(fminf(x0, x1), fminf(x2, x3));
    float gx2 = fmaxf(fmaxf(x0, x1), fmaxf(x2, x3));
    float gy1 = fminf(fminf(y0, y1), fminf(y2, y3));
    float gy2 = fmaxf(fmaxf(y0, y1), fmaxf(y2, y3));
    sgx1[t] = gx1; sgx2[t] = gx2; sgy1[t] = gy1; sgy2[t] = gy2;
    sarea[t] = (gx2 - gx1) * (gy2 - gy1);
    scx[t * 4 + 0] = x0; scy[t * 4 + 0] = y0;   // lt
    scx[t * 4 + 1] = x1; scy[t * 4 + 1] = y1;   // lb
    scx[t * 4 + 2] = x2; scy[t * 4 + 2] = y2;   // rb
    scx[t * 4 + 3] = x3; scy[t * 4 + 3] = y3;   // rt
    slabel[t] = gt_labels[b * NGT + t];
  }
  __syncthreads();

  const int a = blockIdx.x * 256 + t;
  if (a >= NA) return;

  const float* pb = pd_bboxes + ((size_t)b * NA + a) * 9;
  float qx0 = pb[0], qy0 = pb[1], qx1 = pb[2], qy1 = pb[3];
  float qx2 = pb[4], qy2 = pb[5], qx3 = pb[6], qy3 = pb[7];
  float px1 = fminf(fminf(qx0, qx1), fminf(qx2, qx3));
  float px2 = fmaxf(fmaxf(qx0, qx1), fmaxf(qx2, qx3));
  float py1 = fminf(fminf(qy0, qy1), fminf(qy2, qy3));
  float py2 = fmaxf(fmaxf(qy0, qy1), fmaxf(qy2, qy3));
  float area_p = (px2 - px1) * (py2 - py1);
  float ax = anc[a * 2 + 0], ay = anc[a * 2 + 1];
  const float* srow = pd_scores + ((size_t)b * NA + a) * NC;

  for (int g = 0; g < NGT; ++g) {
    float iw = fmaxf(fminf(sgx2[g], px2) - fmaxf(sgx1[g], px1), 0.f);
    float ih = fmaxf(fminf(sgy2[g], py2) - fmaxf(sgy1[g], py1), 0.f);
    float inter = iw * ih;
    float uni = sarea[g] + area_p - inter + EPSF;
    float o = inter / uni;
    if (o < 0.f) o = 0.f;

    float ltx = scx[g*4+0], lty = scy[g*4+0];
    float lbx = scx[g*4+1], lby = scy[g*4+1];
    float rbx = scx[g*4+2], rby = scy[g*4+2];
    float rtx = scx[g*4+3], rty = scy[g*4+3];
    float vwx = rtx - ltx, vwy = rty - lty;
    float vhx = lbx - ltx, vhy = lby - lty;
    float fltx = ax - ltx, flty = ay - lty;
    float flbx = ax - lbx, flby = ay - lby;
    float frbx = ax - rbx, frby = ay - rby;
    float frtx = ax - rtx, frty = ay - rty;
    int res = (int)( (fltx*vwx + flty*vwy)  > EPSF)
            + (int)( (fltx*vhx + flty*vhy)  > EPSF)
            + (int)( (flbx*vwx + flby*vwy)  > EPSF)
            + (int)(-(flbx*vhx + flby*vhy)  > EPSF)
            + (int)(-(frbx*vwx + frby*vwy)  > EPSF)
            + (int)(-(frbx*vhx + frby*vhy)  > EPSF)
            + (int)(-(frtx*vwx + frty*vwy)  > EPSF)
            + (int)( (frtx*vhx + frty*vhy)  > EPSF);
    unsigned char mi = (res >= 8) ? (unsigned char)1 : (unsigned char)0;

    float sc = srow[slabel[g]];
    float o2 = o * o;
    float m = sc * (o2 * o2 * o2) * (float)mi;   // ALPHA=1, BETA=6

    size_t off = ((size_t)b * NGT + g) * NA + a;
    ov[off] = o; met[off] = m; maskin[off] = mi;
  }
}

// ---------------------------------------------------------------------------
// Kernel 2: per (b, gt) iterative top-13 over na=8400 (jax tie-break: lowest
// index among equal values). mask_gt==0 -> sentinel -1 (zeroed row in ref).
// ---------------------------------------------------------------------------
__global__ void k_topk(const float* __restrict__ met,
                       const float* __restrict__ mask_gt,
                       int* __restrict__ topk_idx) {
  const int bg = blockIdx.x;
  const int t  = threadIdx.x;
  __shared__ float sval[256];
  __shared__ int   sidx[256];
  __shared__ int   ssel[TOPKK];

  if (mask_gt[bg] <= 0.f) {
    if (t < TOPKK) topk_idx[bg * TOPKK + t] = -1;
    return;
  }
  const float* row = met + (size_t)bg * NA;

  for (int k = 0; k < TOPKK; ++k) {
    float bv = -1.f; int bi = 0x7fffffff;
    for (int a = t; a < NA; a += 256) {
      bool excl = false;
      for (int q = 0; q < k; ++q) excl |= (ssel[q] == a);
      if (!excl) {
        float v = row[a];
        if (v > bv || (v == bv && a < bi)) { bv = v; bi = a; }
      }
    }
    sval[t] = bv; sidx[t] = bi;
    __syncthreads();
    for (int s = 128; s > 0; s >>= 1) {
      if (t < s) {
        float v2 = sval[t + s]; int i2 = sidx[t + s];
        if (v2 > sval[t] || (v2 == sval[t] && i2 < sidx[t])) { sval[t] = v2; sidx[t] = i2; }
      }
      __syncthreads();
    }
    if (t == 0) { ssel[k] = sidx[0]; topk_idx[bg * TOPKK + k] = sidx[0]; }
    __syncthreads();
  }
}

// ---------------------------------------------------------------------------
// Kernel 3: scatter mask_pos = topk_indicator * mask_in_gts (mask_gt via -1).
// ---------------------------------------------------------------------------
__global__ void k_scatter(const int* __restrict__ topk_idx,
                          const unsigned char* __restrict__ maskin,
                          unsigned char* __restrict__ maskpos) {
  const int bg = blockIdx.x;
  const int t  = threadIdx.x;
  if (t < TOPKK) {
    int a = topk_idx[bg * TOPKK + t];
    if (a >= 0) {
      size_t off = (size_t)bg * NA + a;
      maskpos[off] = maskin[off];
    }
  }
}

// ---------------------------------------------------------------------------
// Kernel 4: WMMA resolve. One wave32 per 16-anchor tile.
//   fg[a] = sum_g mask_pos[g][a] computed as (16x64 f16) x (ones 64x16) via
//   two V_WMMA_F32_16X16X32_F16. Then multi-assignment fixup (argmax overlap)
//   and target_gt_idx extraction.
// ---------------------------------------------------------------------------
__global__ void k_resolve(unsigned char* __restrict__ maskpos,
                          const float* __restrict__ ov,
                          int* __restrict__ fg_out,
                          int* __restrict__ tgi_out) {
  const int tiles = NA / 16;               // 525
  const int b  = blockIdx.x / tiles;
  const int a0 = (blockIdx.x % tiles) * 16;
  const int lane = threadIdx.x;            // 0..31, full wave
  const int M = lane & 15;
  const int h = lane >> 4;

  // Build A tiles (anchors x gts) per the 16-bit A 16x32 VGPR layout:
  // VGPR j(0..3): K = 2j+8h, 2j+1+8h ; VGPR 4..7: K = 16+2(j-4)+8h, +1.
  v16h A0, A1;
  for (int j = 0; j < 8; ++j) {
    int k0 = (j < 4) ? (2 * j + 8 * h) : (16 + 2 * (j - 4) + 8 * h);
    A0[2 * j]     = (_Float16)(float)maskpos[((size_t)b * NGT + k0     ) * NA + a0 + M];
    A0[2 * j + 1] = (_Float16)(float)maskpos[((size_t)b * NGT + k0 + 1 ) * NA + a0 + M];
    A1[2 * j]     = (_Float16)(float)maskpos[((size_t)b * NGT + k0 + 32) * NA + a0 + M];
    A1[2 * j + 1] = (_Float16)(float)maskpos[((size_t)b * NGT + k0 + 33) * NA + a0 + M];
  }
  v16h B;
  for (int j = 0; j < 16; ++j) B[j] = (_Float16)1.0f;   // ones: layout-invariant

  v8f C = {};
  C = __builtin_amdgcn_wmma_f32_16x16x32_f16(false, A0, false, B, (short)0, C, false, false);
  C = __builtin_amdgcn_wmma_f32_16x16x32_f16(false, A1, false, B, (short)0, C, false, false);

  // D layout: lane n<16 VGPR i -> (M=i, N=n); lane n>=16 -> (M=i+8).
  __shared__ int sh_fg[16];
  if ((lane & 15) == 0) {                  // lanes 0 and 16 (column N=0)
    for (int i = 0; i < 8; ++i) sh_fg[h * 8 + i] = (int)(C[i] + 0.5f);
  }
  __syncthreads();

  if (lane < 16) {
    const int a = a0 + lane;
    int fg = sh_fg[lane];
    int tgi = 0, fgf = fg;
    if (fg > 1) {
      // replace column with one-hot of argmax_g overlaps (first max, like jnp)
      float best = -1.f; int amax = 0;
      for (int g = 0; g < NGT; ++g) {
        float o = ov[((size_t)b * NGT + g) * NA + a];
        if (o > best) { best = o; amax = g; }
      }
      for (int g = 0; g < NGT; ++g)
        maskpos[((size_t)b * NGT + g) * NA + a] = (g == amax) ? 1 : 0;
      tgi = amax; fgf = 1;
    } else {
      for (int g = 0; g < NGT; ++g)
        if (maskpos[((size_t)b * NGT + g) * NA + a]) { tgi = g; break; }
    }
    fg_out[b * NA + a]  = fgf;
    tgi_out[b * NA + a] = tgi;
  }
}

// ---------------------------------------------------------------------------
// Kernel 5: per (b, gt): pos_align = max_a(align*mask_pos), pos_over = max_a(ov*mask_pos)
// (raw align recomputed since is_max fixup can land where mask_in_gts==0).
// ---------------------------------------------------------------------------
__global__ void k_posmax(const float* __restrict__ ov,
                         const unsigned char* __restrict__ maskpos,
                         const float* __restrict__ pd_scores,
                         const int* __restrict__ gt_labels,
                         float* __restrict__ pos_align,
                         float* __restrict__ pos_over) {
  const int bg = blockIdx.x;
  const int b  = bg / NGT;
  const int t  = threadIdx.x;
  const int label = gt_labels[bg];
  __shared__ float sa[256], so[256];

  float am = 0.f, om = 0.f;
  const float* orow = ov + (size_t)bg * NA;
  const unsigned char* mrow = maskpos + (size_t)bg * NA;
  for (int a = t; a < NA; a += 256) {
    if (mrow[a]) {
      float o = orow[a];
      float s = pd_scores[((size_t)b * NA + a) * NC + label];
      float o2 = o * o;
      float raw = s * o2 * o2 * o2;
      am = fmaxf(am, raw); om = fmaxf(om, o);
    }
  }
  sa[t] = am; so[t] = om;
  __syncthreads();
  for (int s2 = 128; s2 > 0; s2 >>= 1) {
    if (t < s2) { sa[t] = fmaxf(sa[t], sa[t + s2]); so[t] = fmaxf(so[t], so[t + s2]); }
    __syncthreads();
  }
  if (t == 0) { pos_align[bg] = sa[0]; pos_over[bg] = so[0]; }
}

// ---------------------------------------------------------------------------
// Kernel 6: finalize -> d_out = [labels | bboxes*9 | scores*80 | fg | tgi] (f32)
// ---------------------------------------------------------------------------
__global__ void k_finalize(const float* __restrict__ pd_scores,
                           const float* __restrict__ ov,
                           const int* __restrict__ gt_labels,
                           const float* __restrict__ gt_bboxes,
                           const float* __restrict__ pos_align,
                           const float* __restrict__ pos_over,
                           const int* __restrict__ fg,
                           const int* __restrict__ tgi,
                           float* __restrict__ out) {
  const int b = blockIdx.y;
  const int a = blockIdx.x * 256 + threadIdx.x;
  if (a >= NA) return;
  const int idx = b * NA + a;
  const int g = tgi[idx];
  const int f = fg[idx];
  const int label = gt_labels[b * NGT + g];

  float norm = 0.f;
  if (f > 0) {
    // after resolve mask_pos has exactly one nonzero per fg column -> max at g
    float o = ov[((size_t)b * NGT + g) * NA + a];
    float s = pd_scores[(size_t)idx * NC + label];
    float o2 = o * o;
    float raw = s * o2 * o2 * o2;
    int bg = b * NGT + g;
    norm = raw * pos_over[bg] / (pos_align[bg] + EPSF);
  }

  out[idx] = (float)label;                                        // target_labels
  float* ob = out + (size_t)BS * NA + (size_t)idx * 9;            // target_bboxes
  const float* src = gt_bboxes + ((size_t)b * NGT + g) * 9;
  for (int j = 0; j < 9; ++j) ob[j] = src[j];
  float* os = out + (size_t)BS * NA * 10 + (size_t)idx * NC;      // target_scores
  for (int c = 0; c < NC; ++c) os[c] = 0.f;
  if (f > 0) os[label] = norm;
  out[(size_t)BS * NA * 90 + idx] = (f > 0) ? 1.f : 0.f;          // fg_mask
  out[(size_t)BS * NA * 91 + idx] = (float)g;                     // target_gt_idx
}

// ---------------------------------------------------------------------------
extern "C" void kernel_launch(void* const* d_in, const int* in_sizes, int n_in,
                              void* d_out, int out_size, void* d_ws, size_t ws_size,
                              hipStream_t stream) {
  (void)in_sizes; (void)n_in; (void)out_size; (void)ws_size;
  const float* pd_scores = (const float*)d_in[0];
  const float* pd_bboxes = (const float*)d_in[1];
  const float* anc       = (const float*)d_in[2];
  const int*   gt_labels = (const int*)d_in[3];
  const float* gt_bboxes = (const float*)d_in[4];
  const float* mask_gt   = (const float*)d_in[5];
  float* out = (float*)d_out;

  const size_t S = (size_t)BS * NGT * NA;       // 8,601,600 elements per plane
  char* ws = (char*)d_ws;
  float* ov              = (float*)ws;          ws += S * 4;
  float* met             = (float*)ws;          ws += S * 4;
  unsigned char* maskin  = (unsigned char*)ws;  ws += S;
  unsigned char* maskpos = (unsigned char*)ws;  ws += S;
  int*   topk            = (int*)ws;            ws += (size_t)BS * NGT * TOPKK * 4;
  float* pos_align       = (float*)ws;          ws += (size_t)BS * NGT * 4;
  float* pos_over        = (float*)ws;          ws += (size_t)BS * NGT * 4;
  int*   fgb             = (int*)ws;            ws += (size_t)BS * NA * 4;
  int*   tgib            = (int*)ws;            ws += (size_t)BS * NA * 4;

  hipMemsetAsync(maskpos, 0, S, stream);

  dim3 gridA((NA + 255) / 256, BS);
  k_metrics<<<gridA, 256, 0, stream>>>(pd_scores, pd_bboxes, anc, gt_labels, gt_bboxes,
                                       ov, met, maskin);
  k_topk<<<BS * NGT, 256, 0, stream>>>(met, mask_gt, topk);
  k_scatter<<<BS * NGT, 32, 0, stream>>>(topk, maskin, maskpos);
  k_resolve<<<BS * (NA / 16), 32, 0, stream>>>(maskpos, ov, fgb, tgib);
  k_posmax<<<BS * NGT, 256, 0, stream>>>(ov, maskpos, pd_scores, gt_labels,
                                         pos_align, pos_over);
  k_finalize<<<gridA, 256, 0, stream>>>(pd_scores, ov, gt_labels, gt_bboxes,
                                        pos_align, pos_over, fgb, tgib, out);
}
